// NaiveCompressedLinear_48369921688094
// MI455X (gfx1250) — matrix operand order repaired and checked by reading, not compiled, not run on hardware
//
#include <hip/hip_runtime.h>

// ---------------------------------------------------------------------------
// Quantized linear  y = x @ ((w_q - zp) * scale)^T + bias   (gfx1250 WMMA)
// M = B*S = 8192, N = D_OUT = 4096, K = D_IN = 4096.
//
// Strategy: weights/activations are converted to f16 ONCE into d_ws
// ((w_q - zp) in [-255,255] is exact in f16), then a pure-f16 WMMA GEMM runs
// with zero in-loop dequant VALU. Staging (96 MB) fits the 192 MB L2, so the
// GEMM streams from cache. scale/bias applied in the fp32 epilogue.
// Fallback: fused convert-in-loop kernel if ws_size is too small.
// ---------------------------------------------------------------------------

typedef __attribute__((ext_vector_type(16))) _Float16 v16h;
typedef __attribute__((ext_vector_type(8)))  _Float16 v8h;
typedef __attribute__((ext_vector_type(8)))  float    v8f;
typedef __attribute__((ext_vector_type(4)))  float    f32x4;
typedef __attribute__((ext_vector_type(4)))  int      i32x4;

// ============================ conversion kernels ===========================

__global__ __launch_bounds__(256)
void cvt_x_kernel(const float* __restrict__ X, _Float16* __restrict__ Xh, size_t n)
{
    size_t i = ((size_t)blockIdx.x * blockDim.x + threadIdx.x) * 8;
    if (i >= n) return;
    f32x4 a = *(const f32x4*)(X + i);
    f32x4 b = *(const f32x4*)(X + i + 4);
    v8h h;
    #pragma unroll
    for (int e = 0; e < 4; ++e) { h[e] = (_Float16)a[e]; h[4 + e] = (_Float16)b[e]; }
    *(v8h*)(Xh + i) = h;
}

__global__ __launch_bounds__(256)
void cvt_w_kernel(const int* __restrict__ Wq, const int* __restrict__ zp,
                  _Float16* __restrict__ Wh, int K, size_t n)
{
    size_t i = ((size_t)blockIdx.x * blockDim.x + threadIdx.x) * 8;
    if (i >= n) return;
    const int zpi = zp[i / (size_t)K];     // K % 8 == 0 -> one row per 8-vector
    i32x4 a = *(const i32x4*)(Wq + i);
    i32x4 b = *(const i32x4*)(Wq + i + 4);
    v8h h;
    #pragma unroll
    for (int e = 0; e < 4; ++e) {
        h[e]     = (_Float16)(float)(a[e] - zpi);   // exact: |w-zp| <= 255
        h[4 + e] = (_Float16)(float)(b[e] - zpi);
    }
    *(v8h*)(Wh + i) = h;
}

// ============================== f16 WMMA GEMM ==============================
// Block: 256 threads = 8 waves. Block tile 128(M) x 256(N), BK = 32.
// Wave grid 2(M) x 4(N); each wave computes 64x64 = 4x4 WMMA tiles.

constexpr int GBM = 128, GBN = 256, GBK = 32;
constexpr int GLD = 40;   // padded LDS row stride in halves (80 B, 16B-aligned)

__global__ __launch_bounds__(256)
void gemm_f16_wmma_kernel(const _Float16* __restrict__ Xh,    // [M,K]
                          const _Float16* __restrict__ Wh,    // [N,K]
                          const float*    __restrict__ scale, // [N]
                          const float*    __restrict__ bias,  // [N]
                          float*          __restrict__ Y,     // [M,N]
                          int M, int N, int K)
{
    __shared__ __align__(16) _Float16 As[2][GBM][GLD];
    __shared__ __align__(16) _Float16 Bs[2][GBN][GLD];

    const int tid  = threadIdx.x;
    const int lane = tid & 31;
    const int wave = tid >> 5;
    const int hl   = lane >> 4;
    const int l16  = lane & 15;

    const int blockM = blockIdx.y * GBM;
    const int blockN = blockIdx.x * GBN;

    // A loader: thread t -> row t/2 (0..127), K-halfsegment (t&1)*16
    const int alr = tid >> 1;
    const int alk = (tid & 1) * 16;
    const _Float16* xg = Xh + (size_t)(blockM + alr) * K + alk;
    // B loader: thread t -> row t (0..255), full 32-half K segment
    const _Float16* wg = Wh + (size_t)(blockN + tid) * K;

    const int wm = wave & 1;   // 0..1  (64 rows each)
    const int wn = wave >> 1;  // 0..3  (64 cols each)

    v8f acc[4][4];
    #pragma unroll
    for (int mt = 0; mt < 4; ++mt)
        #pragma unroll
        for (int nt = 0; nt < 4; ++nt)
            #pragma unroll
            for (int v = 0; v < 8; ++v)
                acc[mt][nt][v] = 0.0f;

    v8h xa[2], wb[4];

    auto gload = [&](int kk) {
        const _Float16* xp = xg + (size_t)kk * GBK;
        const _Float16* wp = wg + (size_t)kk * GBK;
        #pragma unroll
        for (int j = 0; j < 2; ++j) xa[j] = *(const v8h*)(xp + j * 8);
        #pragma unroll
        for (int j = 0; j < 4; ++j) wb[j] = *(const v8h*)(wp + j * 8);
    };

    auto lstore = [&](int buf) {
        #pragma unroll
        for (int j = 0; j < 2; ++j) *(v8h*)&As[buf][alr][alk + j * 8] = xa[j];
        #pragma unroll
        for (int j = 0; j < 4; ++j) *(v8h*)&Bs[buf][tid][j * 8] = wb[j];
    };

    auto compute = [&](int buf) {
        v16h af[4], bf[4];
        // A 16x32 f16 fragment: lane = row, halves 0..7 = K@hl*8, 8..15 = K@16+hl*8
        #pragma unroll
        for (int mt = 0; mt < 4; ++mt) {
            const _Float16* rp = &As[buf][wm * 64 + mt * 16 + l16][0];
            v8h lo = *(const v8h*)(rp + hl * 8);
            v8h hi = *(const v8h*)(rp + 16 + hl * 8);
            af[mt] = __builtin_shufflevector(lo, hi,
                     0,1,2,3,4,5,6,7,8,9,10,11,12,13,14,15);
        }
        // B 32x16 f16 fragment: lane = column, halves 0..7 = K@hl*8, 8..15 = K@16+hl*8
        #pragma unroll
        for (int nt = 0; nt < 4; ++nt) {
            const _Float16* rp = &Bs[buf][wn * 64 + nt * 16 + l16][0];
            v8h lo = *(const v8h*)(rp + hl * 8);
            v8h hi = *(const v8h*)(rp + 16 + hl * 8);
            bf[nt] = __builtin_shufflevector(lo, hi,
                     0,1,2,3,4,5,6,7,8,9,10,11,12,13,14,15);
        }
        #pragma unroll
        for (int mt = 0; mt < 4; ++mt)
            #pragma unroll
            for (int nt = 0; nt < 4; ++nt)
                acc[mt][nt] = __builtin_amdgcn_wmma_f32_16x16x32_f16(
                    false, af[mt], false, bf[nt], (short)0, acc[mt][nt],
                    false, false);
    };

    const int NK = K / GBK;

    gload(0);
    lstore(0);
    __syncthreads();

    for (int kk = 0; kk < NK; ++kk) {
        if (kk + 1 < NK) gload(kk + 1);      // hide global latency under WMMAs
        compute(kk & 1);
        if (kk + 1 < NK) lstore((kk + 1) & 1);
        __syncthreads();
    }

    // Epilogue: y = acc * scale[col] + bias[col]
    // C/D layout: lane%16 = N column, VGPR v -> M row = v + 8*(lane/16)
    #pragma unroll
    for (int nt = 0; nt < 4; ++nt) {
        const int col = blockN + wn * 64 + nt * 16 + l16;
        const float sc = scale[col];
        const float bi = bias[col];
        #pragma unroll
        for (int mt = 0; mt < 4; ++mt) {
            const int row0 = blockM + wm * 64 + mt * 16 + hl * 8;
            #pragma unroll
            for (int v = 0; v < 8; ++v)
                Y[(size_t)(row0 + v) * N + col] = acc[mt][nt][v] * sc + bi;
        }
    }
}

// ===================== fused fallback (ws too small) =======================

constexpr int BM = 128, BN = 128, BK = 32, LDA = 40;

__global__ __launch_bounds__(256)
void qlinear_fused_kernel(const float* __restrict__ X, const int* __restrict__ Wq,
                          const float* __restrict__ scale, const int* __restrict__ zp,
                          const float* __restrict__ bias, float* __restrict__ Y,
                          int M, int N, int K)
{
    __shared__ __align__(16) _Float16 As[2][BM][LDA];
    __shared__ __align__(16) _Float16 Bs[2][BN][LDA];

    const int tid  = threadIdx.x;
    const int lane = tid & 31;
    const int wave = tid >> 5;
    const int hl   = lane >> 4;
    const int l16  = lane & 15;

    const int blockM = blockIdx.y * BM;
    const int blockN = blockIdx.x * BN;

    const int lr = tid >> 1;
    const int lk = (tid & 1) * 16;

    const float* xg  = X  + (size_t)(blockM + lr) * K + lk;
    const int*   wg  = Wq + (size_t)(blockN + lr) * K + lk;
    const float  zpf = (float)zp[blockN + lr];

    const int wm = wave & 1;
    const int wn = wave >> 1;

    v8f acc[4][2];
    #pragma unroll
    for (int mt = 0; mt < 4; ++mt)
        #pragma unroll
        for (int nt = 0; nt < 2; ++nt)
            #pragma unroll
            for (int v = 0; v < 8; ++v)
                acc[mt][nt][v] = 0.0f;

    f32x4 xr[4];
    i32x4 wr[4];

    auto gload = [&](int kk) {
        const float* xp = xg + (size_t)kk * BK;
        const int*   wp = wg + (size_t)kk * BK;
        #pragma unroll
        for (int j = 0; j < 4; ++j) {
            xr[j] = *(const f32x4*)(xp + j * 4);
            wr[j] = *(const i32x4*)(wp + j * 4);
        }
    };

    auto lstore = [&](int buf) {
        v8h ha0, ha1, hb0, hb1;
        #pragma unroll
        for (int j = 0; j < 2; ++j)
            #pragma unroll
            for (int e = 0; e < 4; ++e) {
                ha0[j * 4 + e] = (_Float16)xr[j][e];
                ha1[j * 4 + e] = (_Float16)xr[j + 2][e];
                hb0[j * 4 + e] = (_Float16)((float)wr[j][e]     - zpf);
                hb1[j * 4 + e] = (_Float16)((float)wr[j + 2][e] - zpf);
            }
        *(v8h*)&As[buf][lr][lk]     = ha0;
        *(v8h*)&As[buf][lr][lk + 8] = ha1;
        *(v8h*)&Bs[buf][lr][lk]     = hb0;
        *(v8h*)&Bs[buf][lr][lk + 8] = hb1;
    };

    auto compute = [&](int buf) {
        v16h af[4], bf[2];
        #pragma unroll
        for (int mt = 0; mt < 4; ++mt) {
            const _Float16* rp = &As[buf][wm * 64 + mt * 16 + l16][0];
            v8h lo = *(const v8h*)(rp + hl * 8);
            v8h hi = *(const v8h*)(rp + 16 + hl * 8);
            af[mt] = __builtin_shufflevector(lo, hi,
                     0,1,2,3,4,5,6,7,8,9,10,11,12,13,14,15);
        }
        #pragma unroll
        for (int nt = 0; nt < 2; ++nt) {
            const _Float16* rp = &Bs[buf][wn * 32 + nt * 16 + l16][0];
            v8h lo = *(const v8h*)(rp + hl * 8);
            v8h hi = *(const v8h*)(rp + 16 + hl * 8);
            bf[nt] = __builtin_shufflevector(lo, hi,
                     0,1,2,3,4,5,6,7,8,9,10,11,12,13,14,15);
        }
        #pragma unroll
        for (int mt = 0; mt < 4; ++mt)
            #pragma unroll
            for (int nt = 0; nt < 2; ++nt)
                acc[mt][nt] = __builtin_amdgcn_wmma_f32_16x16x32_f16(
                    false, af[mt], false, bf[nt], (short)0, acc[mt][nt],
                    false, false);
    };

    const int NK = K / BK;
    gload(0);
    lstore(0);
    __syncthreads();
    for (int kk = 0; kk < NK; ++kk) {
        if (kk + 1 < NK) gload(kk + 1);
        compute(kk & 1);
        if (kk + 1 < NK) lstore((kk + 1) & 1);
        __syncthreads();
    }

    #pragma unroll
    for (int nt = 0; nt < 2; ++nt) {
        const int col = blockN + wn * 32 + nt * 16 + l16;
        const float sc = scale[col];
        const float bi = bias[col];
        #pragma unroll
        for (int mt = 0; mt < 4; ++mt) {
            const int row0 = blockM + wm * 64 + mt * 16 + hl * 8;
            #pragma unroll
            for (int v = 0; v < 8; ++v)
                Y[(size_t)(row0 + v) * N + col] = acc[mt][nt][v] * sc + bi;
        }
    }
}

// ================================ launcher =================================

extern "C" void kernel_launch(void* const* d_in, const int* in_sizes, int n_in,
                              void* d_out, int out_size, void* d_ws, size_t ws_size,
                              hipStream_t stream) {
    (void)n_in; (void)out_size;

    const float* X     = (const float*)d_in[0];
    const int*   Wq    = (const int*)  d_in[1];
    const float* scale = (const float*)d_in[2];
    const int*   zp    = (const int*)  d_in[3];
    const float* bias  = (const float*)d_in[4];
    float*       Y     = (float*)d_out;

    const int D_OUT = in_sizes[2];
    const int D_IN  = in_sizes[1] / D_OUT;
    const int Mrows = in_sizes[0] / D_IN;

    const size_t xElems = (size_t)in_sizes[0];
    const size_t wElems = (size_t)in_sizes[1];
    const size_t need   = (xElems + wElems) * 2;   // f16 staging

    const bool shapesOk = (Mrows % GBM == 0) && (D_OUT % GBN == 0) &&
                          (D_IN % GBK == 0) && (D_IN % 8 == 0);

    if (ws_size >= need && shapesOk) {
        _Float16* Xh = (_Float16*)d_ws;
        _Float16* Wh = (_Float16*)((char*)d_ws + xElems * 2);

        const int cvtThreads = 256;
        dim3 gx((unsigned)((xElems / 8 + cvtThreads - 1) / cvtThreads));
        dim3 gw((unsigned)((wElems / 8 + cvtThreads - 1) / cvtThreads));
        cvt_x_kernel<<<gx, cvtThreads, 0, stream>>>(X, Xh, xElems);
        cvt_w_kernel<<<gw, cvtThreads, 0, stream>>>(Wq, zp, Wh, D_IN, wElems);

        dim3 grid(D_OUT / GBN, Mrows / GBM);
        gemm_f16_wmma_kernel<<<grid, 256, 0, stream>>>(Xh, Wh, scale, bias, Y,
                                                       Mrows, D_OUT, D_IN);
    } else {
        dim3 grid(D_OUT / BN, Mrows / BM);
        qlinear_fused_kernel<<<grid, 256, 0, stream>>>(X, Wq, scale, zp, bias, Y,
                                                       Mrows, D_OUT, D_IN);
    }
}
